// siameseRPN_one_branch_29317446762595
// MI455X (gfx1250) — matrix operand order repaired and checked by reading, not compiled, not run on hardware
//
#include <hip/hip_runtime.h>

typedef float  v8f   __attribute__((ext_vector_type(8)));
typedef __bf16 v16bf __attribute__((ext_vector_type(16)));
typedef unsigned int u32x4 __attribute__((ext_vector_type(4)));
typedef int          i32x4 __attribute__((ext_vector_type(4)));
typedef int          i32x8 __attribute__((ext_vector_type(8)));

#define CIN     512
#define KDIM    4608      // 512 * 3 * 3
#define KT_     144       // KDIM / 32 (k-tiles of 32 for bf16 WMMA)
#define NPIX    1024      // 32 * 32
#define EH      128       // EXPAND(64) * 2 halves of CORR
#define MT_TGT  16        // 256 out channels / 16
#define NT_TGT  64        // 1024 pixels / 16
#define MT_TPL  8         // 128 merged rows / 16
#define NT_TPL  7         // ceil(2*49 / 16)

#if __has_builtin(__builtin_amdgcn_tensor_load_to_lds) && \
    __has_builtin(__builtin_amdgcn_s_wait_tensorcnt)
#define HAVE_TDM 1
#else
#define HAVE_TDM 0
#endif

__device__ __forceinline__ unsigned short f2bf(float f) {
  union { float f; unsigned u; } v; v.f = f;
  unsigned u = v.u;
  u += 0x7FFFu + ((u >> 16) & 1u);          // round-to-nearest-even
  return (unsigned short)(u >> 16);
}

#if HAVE_TDM
// TDM descriptor for a contiguous 32 KB global -> LDS copy (1-D tile of 8192
// 4-byte elements). Bitfields per CDNA5 ISA:
//   group0: [1:0]=count=1, [63:32]=lds_addr, [120:64]=global_addr, [127:126]=type=2
//   group1: [17:16]=data_size(2 -> 4B), [79:48]=tensor_dim0=8192,
//           [111:80]=tensor_dim1=1, [127:112]=tile_dim0=8192,
//           [143:128]=tile_dim1=1, [207:160]=tensor_dim0_stride=8192
__device__ __forceinline__ void tdm_copy_32k(const void* gptr, unsigned lds_off) {
  unsigned long long ga = (unsigned long long)gptr;
  u32x4 g0;
  g0[0] = 1u;                                             // count=1, user desc
  g0[1] = lds_off;                                        // lds_addr (bytes)
  g0[2] = (unsigned)(ga & 0xFFFFFFFFu);                   // global_addr[31:0]
  g0[3] = (unsigned)((ga >> 32) & 0x01FFFFFFu) | 0x80000000u;  // [56:32] | type=2
  i32x8 g1;
  g1[0] = 2 << 16;                    // workgroup_mask=0, data_size=4B
  g1[1] = (int)(8192u << 16);         // tensor_dim0[15:0] @ bits 63:48
  g1[2] = 1 << 16;                    // tensor_dim0 hi=0, tensor_dim1=1
  g1[3] = (int)(8192u << 16);         // tile_dim0=8192 @ bits 127:112
  g1[4] = 1;                          // tile_dim1=1, tile_dim2=0
  g1[5] = 8192;                       // tensor_dim0_stride lo
  g1[6] = 0;
  g1[7] = 0;
  i32x4 gz = {0, 0, 0, 0};
#if __clang_major__ >= 23
  i32x8 gz8 = {0, 0, 0, 0, 0, 0, 0, 0};
  __builtin_amdgcn_tensor_load_to_lds(g0, g1, gz, gz, gz8, 0);
#else
  __builtin_amdgcn_tensor_load_to_lds(g0, g1, gz, gz, 0);
#endif
}
#endif  // HAVE_TDM

// 16-bit fragment layouts used here (same k order in A and B) are contiguous K runs
// per lane:
//   A (16x32, MxK): lane -> m = lane&15; k run  [16*hi, 16*hi+16).
//   B (32x16, KxN): lane -> n = lane&15; k run  [16*hi, 16*hi+16).
// so each pack thread emits one full 32-byte lane row (two uint4 stores).

// Pack a row-major f32 [M][KDIM] weight matrix into bf16 WMMA A-fragments.
__global__ void pack_a_frag(const float* __restrict__ src,
                            unsigned short* __restrict__ dst, int Mtiles) {
  int idx = blockIdx.x * blockDim.x + threadIdx.x;    // (kt*Mtiles+mt)*32+lane
  if (idx >= KT_ * Mtiles * 32) return;
  int lane = idx & 31;
  int mtkt = idx >> 5;
  int mt = mtkt % Mtiles;
  int kt = mtkt / Mtiles;
  int m = mt * 16 + (lane & 15);
  const float* row = src + (size_t)m * KDIM + kt * 32 + 16 * (lane >> 4);
  unsigned short o[16];
  #pragma unroll
  for (int s = 0; s < 8; ++s) o[s] = f2bf(row[s]);
  #pragma unroll
  for (int s = 0; s < 8; ++s) o[8 + s] = f2bf(row[8 + s]);
  uint4* d = (uint4*)(dst + (size_t)idx * 16);
  d[0] = *(const uint4*)o;
  d[1] = *(const uint4*)(o + 8);
}

// im2col of target_feat (512,32,32), 3x3 pad 1, into bf16 WMMA B-fragments.
__global__ void pack_b_target(const float* __restrict__ tf,
                              unsigned short* __restrict__ dst) {
  int idx = blockIdx.x * blockDim.x + threadIdx.x;    // (kt*64+nt)*32+lane
  if (idx >= KT_ * NT_TGT * 32) return;
  int lane = idx & 31;
  int nt = (idx >> 5) & 63;
  int kt = idx >> 11;
  int n = nt * 16 + (lane & 15);                      // pixel = y*32 + x
  int y = n >> 5, x = n & 31;
  int k0 = kt * 32 + 16 * (lane >> 4);
  unsigned short o[16];
  #pragma unroll
  for (int s = 0; s < 16; ++s) {
    int k = k0 + s;
    int ci = k / 9, r9 = k % 9;
    int iy = y + r9 / 3 - 1, ix = x + r9 % 3 - 1;
    float v = 0.f;
    if ((unsigned)iy < 32u && (unsigned)ix < 32u)
      v = tf[((size_t)ci * 32 + iy) * 32 + ix];
    o[s] = f2bf(v);
  }
  uint4* d = (uint4*)(dst + (size_t)idx * 16);
  d[0] = *(const uint4*)o;
  d[1] = *(const uint4*)(o + 8);
}

// im2col of template_feat (2,512,7,7), 3x3 pad 1. N = n_tpl*49 + pixel, pad to 112.
__global__ void pack_b_template(const float* __restrict__ tf,
                                unsigned short* __restrict__ dst) {
  int idx = blockIdx.x * blockDim.x + threadIdx.x;
  if (idx >= KT_ * NT_TPL * 32) return;
  int lane = idx & 31;
  int nt = (idx >> 5) % NT_TPL;
  int kt = (idx >> 5) / NT_TPL;
  int n_lin = nt * 16 + (lane & 15);
  int k0 = kt * 32 + 16 * (lane >> 4);
  bool valid = n_lin < 98;
  int ntpl = 0, py = 0, px = 0;
  if (valid) {
    ntpl = n_lin / 49;
    int p = n_lin % 49;
    py = p / 7; px = p % 7;
  }
  unsigned short o[16];
  #pragma unroll
  for (int s = 0; s < 16; ++s) {
    int k = k0 + s;
    int ci = k / 9, r9 = k % 9;
    int iy = py + r9 / 3 - 1, ix = px + r9 % 3 - 1;
    float v = 0.f;
    if (valid && (unsigned)iy < 7u && (unsigned)ix < 7u)
      v = tf[(((size_t)ntpl * CIN + ci) * 7 + iy) * 7 + ix];
    o[s] = f2bf(v);
  }
  uint4* d = (uint4*)(dst + (size_t)idx * 16);
  d[0] = *(const uint4*)o;
  d[1] = *(const uint4*)(o + 8);
}

// Fold the 1x1 merge into the template conv weights (float4-vectorized):
// wm[eh][j] = sum_{c'=0..127} w_merge[(eh&1)*128+c'] * w_template[eh*128+c'][j]
// Streams the 302 MB w_template exactly once -> HBM roofline floor (~13 us).
__global__ void merge_wm(const float* __restrict__ w_template,
                         const float* __restrict__ w_merge,
                         float4* __restrict__ wm) {
  int idx = blockIdx.x * blockDim.x + threadIdx.x;    // over EH * KDIM/4
  if (idx >= EH * (KDIM / 4)) return;
  int j4 = idx % (KDIM / 4);
  int eh = idx / (KDIM / 4);
  const float4* wrow = (const float4*)(w_template + (size_t)eh * 128 * KDIM) + j4;
  const float* mw = w_merge + (eh & 1) * 128;
  for (int c = 0; c < 128; c += 32)
    __builtin_prefetch(wrow + (size_t)c * (KDIM / 4), 0, 1);
  float4 acc = {0.f, 0.f, 0.f, 0.f};
  #pragma unroll 4
  for (int c = 0; c < 128; ++c) {
    float w = mw[c];
    float4 v = wrow[(size_t)c * (KDIM / 4)];
    acc.x = fmaf(w, v.x, acc.x);
    acc.y = fmaf(w, v.y, acc.y);
    acc.z = fmaf(w, v.z, acc.z);
    acc.w = fmaf(w, v.w, acc.w);
  }
  wm[idx] = acc;
}

__global__ void merge_bm(const float* __restrict__ b_template,
                         const float* __restrict__ w_merge,
                         float* __restrict__ bm) {
  int eh = blockIdx.x * blockDim.x + threadIdx.x;
  if (eh >= EH) return;
  float acc = 0.f;
  for (int c = 0; c < 128; ++c)
    acc = fmaf(w_merge[(eh & 1) * 128 + c], b_template[eh * 128 + c], acc);
  bm[eh] = acc;
}

// Target conv as implicit GEMM: M=256, N=1024, K=4608.
// Block = 8 waves = one M-tile x 32 N-tiles (half the N dim). Per k-step the
// block consumes one contiguous 32 KB slice of packed B. With TDM available,
// wave 0 streams slice kt+1 into the alternate LDS buffer asynchronously
// (TENSORcnt) while all 8 waves run WMMAs on slice kt out of LDS; A fragments
// stay register-double-buffered from global. Epilogue fuses conv bias + BN.
__global__ void __launch_bounds__(256) gemm_target(
    const unsigned short* __restrict__ Af, const unsigned short* __restrict__ Bf,
    const float* __restrict__ bias, const float* __restrict__ gamma,
    const float* __restrict__ beta, const float* __restrict__ mean,
    const float* __restrict__ var, float* __restrict__ t_out) {
  int tid = threadIdx.x;
  int wave = tid >> 5;                 // 0..7
  int lane = tid & 31;
  int mt = blockIdx.x >> 1;            // 16 M-tiles
  int nhalf = blockIdx.x & 1;          // which 32-tile half of N
  int ng = nhalf * 8 + wave;           // this wave's group of 4 N-tiles
  const unsigned short* ap = Af + ((size_t)mt * 32 + lane) * 16;
  v8f c[4] = {};

#if HAVE_TDM
  __shared__ unsigned short Bs[2][32 * 512];       // 2 x 32 KB slices
  const unsigned short* bslice = Bf + (size_t)nhalf * (32 * 512);
  if (wave == 0)
    tdm_copy_32k(bslice, (unsigned)(unsigned long long)(uintptr_t)&Bs[0][0]);
  v16bf a = *(const v16bf*)ap;
  for (int kt = 0; kt < KT_; ++kt) {
    if (wave == 0) __builtin_amdgcn_s_wait_tensorcnt(0);   // slice kt landed
    __syncthreads();                                       // publish slice kt
    if (wave == 0 && kt + 1 < KT_)
      tdm_copy_32k(bslice + (size_t)(kt + 1) * (NT_TGT * 512),
                   (unsigned)(unsigned long long)(uintptr_t)&Bs[(kt + 1) & 1][0]);
    int knext = (kt + 1 < KT_) ? kt + 1 : kt;
    v16bf an = *(const v16bf*)(ap + (size_t)knext * (MT_TGT * 512));
    const unsigned short* bb = &Bs[kt & 1][((wave * 4) * 32 + lane) * 16];
    v16bf b0 = *(const v16bf*)(bb);
    v16bf b1 = *(const v16bf*)(bb + 512);
    v16bf b2 = *(const v16bf*)(bb + 1024);
    v16bf b3 = *(const v16bf*)(bb + 1536);
    c[0] = __builtin_amdgcn_wmma_f32_16x16x32_bf16(false, a, false, b0, (short)0, c[0], false, false);
    c[1] = __builtin_amdgcn_wmma_f32_16x16x32_bf16(false, a, false, b1, (short)0, c[1], false, false);
    c[2] = __builtin_amdgcn_wmma_f32_16x16x32_bf16(false, a, false, b2, (short)0, c[2], false, false);
    c[3] = __builtin_amdgcn_wmma_f32_16x16x32_bf16(false, a, false, b3, (short)0, c[3], false, false);
    a = an;
  }
#else
  // Fallback: register-double-buffered global loads.
  const unsigned short* bp = Bf + ((size_t)(ng * 4) * 32 + lane) * 16;
  v16bf a  = *(const v16bf*)ap;
  v16bf b0 = *(const v16bf*)(bp);
  v16bf b1 = *(const v16bf*)(bp + 512);
  v16bf b2 = *(const v16bf*)(bp + 1024);
  v16bf b3 = *(const v16bf*)(bp + 1536);
  for (int kt = 1; kt < KT_; ++kt) {
    const unsigned short* apn = ap + (size_t)kt * (MT_TGT * 512);
    const unsigned short* bpn = bp + (size_t)kt * (NT_TGT * 512);
    v16bf an  = *(const v16bf*)apn;
    v16bf bn0 = *(const v16bf*)(bpn);
    v16bf bn1 = *(const v16bf*)(bpn + 512);
    v16bf bn2 = *(const v16bf*)(bpn + 1024);
    v16bf bn3 = *(const v16bf*)(bpn + 1536);
    c[0] = __builtin_amdgcn_wmma_f32_16x16x32_bf16(false, a, false, b0, (short)0, c[0], false, false);
    c[1] = __builtin_amdgcn_wmma_f32_16x16x32_bf16(false, a, false, b1, (short)0, c[1], false, false);
    c[2] = __builtin_amdgcn_wmma_f32_16x16x32_bf16(false, a, false, b2, (short)0, c[2], false, false);
    c[3] = __builtin_amdgcn_wmma_f32_16x16x32_bf16(false, a, false, b3, (short)0, c[3], false, false);
    a = an; b0 = bn0; b1 = bn1; b2 = bn2; b3 = bn3;
  }
  c[0] = __builtin_amdgcn_wmma_f32_16x16x32_bf16(false, a, false, b0, (short)0, c[0], false, false);
  c[1] = __builtin_amdgcn_wmma_f32_16x16x32_bf16(false, a, false, b1, (short)0, c[1], false, false);
  c[2] = __builtin_amdgcn_wmma_f32_16x16x32_bf16(false, a, false, b2, (short)0, c[2], false, false);
  c[3] = __builtin_amdgcn_wmma_f32_16x16x32_bf16(false, a, false, b3, (short)0, c[3], false, false);
#endif

  int hi = lane >> 4, ln = lane & 15;
  #pragma unroll
  for (int r = 0; r < 8; ++r) {
    int m = mt * 16 + r + 8 * hi;            // ISA C-layout: VGPR r -> M=r(+8)
    float sc = gamma[m] * rsqrtf(var[m] + 1e-5f);
    float off = beta[m] - mean[m] * sc;
    float bz = bias[m];
    #pragma unroll
    for (int j = 0; j < 4; ++j) {
      int n = (ng * 4 + j) * 16 + ln;
      t_out[(size_t)m * NPIX + n] = (c[j][r] + bz) * sc + off;
    }
  }
}

// Mini template conv with merged weights: M=128, N=112(98 valid), K=4608.
// Register double-buffer pipeline.
__global__ void __launch_bounds__(256) gemm_template_k(
    const unsigned short* __restrict__ Af, const unsigned short* __restrict__ Bf,
    const float* __restrict__ bm, float* __restrict__ zw) {
  int wave = (blockIdx.x * blockDim.x + threadIdx.x) >> 5;   // 56 waves exactly
  int lane = threadIdx.x & 31;
  int mt = wave / NT_TPL;
  int nt = wave % NT_TPL;
  const unsigned short* ap = Af + ((size_t)mt * 32 + lane) * 16;
  const unsigned short* bp = Bf + ((size_t)nt * 32 + lane) * 16;
  v8f c = {};
  v16bf a = *(const v16bf*)ap;
  v16bf b = *(const v16bf*)bp;
  for (int kt = 1; kt < KT_; ++kt) {
    v16bf an = *(const v16bf*)(ap + (size_t)kt * (MT_TPL * 512));
    v16bf bn = *(const v16bf*)(bp + (size_t)kt * (NT_TPL * 512));
    c = __builtin_amdgcn_wmma_f32_16x16x32_bf16(false, a, false, b, (short)0, c, false, false);
    a = an; b = bn;
  }
  c = __builtin_amdgcn_wmma_f32_16x16x32_bf16(false, a, false, b, (short)0, c, false, false);

  int hi = lane >> 4, ln = lane & 15;
  int n_lin = nt * 16 + ln;
  if (n_lin < 98) {
    int ntpl = n_lin / 49, p = n_lin % 49;
    #pragma unroll
    for (int r = 0; r < 8; ++r) {
      int m = mt * 16 + r + 8 * hi;
      zw[((size_t)ntpl * EH + m) * 49 + p] = c[r] + bm[m];
    }
  }
}

// Fused depthwise correlation + (already folded) 1x1 merge + corr_bias.
// Block = one output plane a x 256 pixels; kernel taps and zero-padded t halo
// staged in LDS so the 98-MAC inner loop has no row bounds checks.
__global__ void __launch_bounds__(256) corr_merge(
    const float* __restrict__ t, const float* __restrict__ zw,
    const float* __restrict__ corr_bias, float* __restrict__ rpn) {
  __shared__ float szw[2][49];
  __shared__ float st[2][14][32];
  int a = blockIdx.x >> 2;                  // 128 planes x 4 blocks each
  int pixbase = (blockIdx.x & 3) * 256;
  int tid = threadIdx.x;
  int n = a >> 6, e = a & 63;
  if (tid < 98) {
    int half = tid / 49, p = tid % 49;
    szw[half][p] = zw[((size_t)n * EH + e * 2 + half) * 49 + p];
  }
  int y0 = pixbase >> 5;                    // first output row of this block
  for (int i = tid; i < 2 * 14 * 32; i += 256) {
    int half = i / (14 * 32);
    int rr = (i >> 5) % 14;
    int xx = i & 31;
    int iy = y0 - 3 + rr;
    float v = 0.f;
    if ((unsigned)iy < 32u)
      v = t[(size_t)(2 * a + half) * NPIX + iy * 32 + xx];
    st[half][rr][xx] = v;
  }
  __syncthreads();
  int pix = pixbase + tid;
  int x = pix & 31;
  int ry = tid >> 5;                        // y - y0
  float acc = 0.f;
  #pragma unroll
  for (int half = 0; half < 2; ++half) {
    #pragma unroll
    for (int dy = 0; dy < 7; ++dy) {
      const float* trow = &st[half][ry + dy][0];
      const float* zrow = &szw[half][dy * 7];
      #pragma unroll
      for (int dx = 0; dx < 7; ++dx) {
        int ix = x + dx - 3;
        if ((unsigned)ix < 32u) acc = fmaf(trow[ix], zrow[dx], acc);
      }
    }
  }
  rpn[(size_t)a * NPIX + pix] = 0.1f * acc + corr_bias[e];
}

// 1x1 cls/box heads + paired softmax over (anchor, anchor+9).
__global__ void heads_softmax(const float* __restrict__ rpn,
                              const float* __restrict__ w_cls, const float* __restrict__ b_cls,
                              const float* __restrict__ w_box, const float* __restrict__ b_box,
                              float* __restrict__ out) {
  int idx = blockIdx.x * blockDim.x + threadIdx.x;
  if (idx >= 2 * NPIX) return;
  int pix = idx & (NPIX - 1);
  int n = idx >> 10;
  float f[64];
  #pragma unroll
  for (int e = 0; e < 64; ++e)
    f[e] = rpn[(size_t)(n * 64 + e) * NPIX + pix];
  float cls[18];
  #pragma unroll
  for (int o = 0; o < 18; ++o) {
    float acc = b_cls[o];
    const float* w = w_cls + o * 64;
    #pragma unroll
    for (int e = 0; e < 64; ++e) acc = fmaf(w[e], f[e], acc);
    cls[o] = acc;
  }
  float* prob_out = out;                   // (2,18,32,32)
  float* box_out = out + 2 * 18 * NPIX;    // (2,36,32,32)
  #pragma unroll
  for (int a9 = 0; a9 < 9; ++a9) {
    float p0 = cls[a9], p1 = cls[a9 + 9];
    float mx = fmaxf(p0, p1);
    float e0 = __expf(p0 - mx), e1 = __expf(p1 - mx);
    float inv = 1.f / (e0 + e1);
    prob_out[((size_t)n * 18 + a9) * NPIX + pix] = e0 * inv;
    prob_out[((size_t)n * 18 + a9 + 9) * NPIX + pix] = e1 * inv;
  }
  #pragma unroll
  for (int o = 0; o < 36; ++o) {
    float acc = b_box[o];
    const float* w = w_box + o * 64;
    #pragma unroll
    for (int e = 0; e < 64; ++e) acc = fmaf(w[e], f[e], acc);
    box_out[((size_t)n * 36 + o) * NPIX + pix] = acc;
  }
}

extern "C" void kernel_launch(void* const* d_in, const int* in_sizes, int n_in,
                              void* d_out, int out_size, void* d_ws, size_t ws_size,
                              hipStream_t stream) {
  (void)in_sizes; (void)n_in; (void)out_size; (void)ws_size;
  const float* target_feat   = (const float*)d_in[0];
  const float* template_feat = (const float*)d_in[1];
  const float* w_target      = (const float*)d_in[2];
  const float* b_target      = (const float*)d_in[3];
  const float* bn_gamma      = (const float*)d_in[4];
  const float* bn_beta       = (const float*)d_in[5];
  const float* bn_mean       = (const float*)d_in[6];
  const float* bn_var        = (const float*)d_in[7];
  const float* w_template    = (const float*)d_in[8];
  const float* b_template    = (const float*)d_in[9];
  const float* w_merge       = (const float*)d_in[10];
  const float* corr_bias     = (const float*)d_in[11];
  const float* w_cls         = (const float*)d_in[12];
  const float* b_cls         = (const float*)d_in[13];
  const float* w_box         = (const float*)d_in[14];
  const float* b_box         = (const float*)d_in[15];
  float* out = (float*)d_out;

  char* ws = (char*)d_ws;
  size_t off = 0;
  auto alloc = [&](size_t bytes) {
    char* p = ws + off;
    off = (off + bytes + 255) & ~(size_t)255;
    return p;
  };
  unsigned short* Af1 = (unsigned short*)alloc((size_t)256 * KDIM * 2);   // 2.36 MB
  unsigned short* Bf1 = (unsigned short*)alloc((size_t)KDIM * NPIX * 2);  // 9.44 MB
  float*          T   = (float*)alloc((size_t)256 * NPIX * 4);            // 1 MB
  float*          WM  = (float*)alloc((size_t)EH * KDIM * 4);             // 2.36 MB
  unsigned short* Af2 = (unsigned short*)alloc((size_t)EH * KDIM * 2);    // 1.18 MB
  unsigned short* Bf2 = (unsigned short*)alloc((size_t)KDIM * 112 * 2);   // 1.03 MB
  float*          ZW  = (float*)alloc((size_t)2 * EH * 49 * 4);
  float*          BM  = (float*)alloc((size_t)EH * 4);
  float*          RPN = (float*)alloc((size_t)EH * NPIX * 4);             // 0.5 MB

  const int B = 256;
  auto gs = [](int nthreads, int b) { return (nthreads + b - 1) / b; };

  pack_a_frag<<<gs(KT_ * MT_TGT * 32, B), B, 0, stream>>>(w_target, Af1, MT_TGT);
  pack_b_target<<<gs(KT_ * NT_TGT * 32, B), B, 0, stream>>>(target_feat, Bf1);
  gemm_target<<<32, B, 0, stream>>>(Af1, Bf1, b_target, bn_gamma, bn_beta,
                                    bn_mean, bn_var, T);
  merge_wm<<<gs(EH * (KDIM / 4), B), B, 0, stream>>>(w_template, w_merge, (float4*)WM);
  merge_bm<<<1, 128, 0, stream>>>(b_template, w_merge, BM);
  pack_a_frag<<<gs(KT_ * MT_TPL * 32, B), B, 0, stream>>>(WM, Af2, MT_TPL);
  pack_b_template<<<gs(KT_ * NT_TPL * 32, B), B, 0, stream>>>(template_feat, Bf2);
  gemm_template_k<<<7, B, 0, stream>>>(Af2, Bf2, BM, ZW);
  corr_merge<<<EH * 4, B, 0, stream>>>(T, ZW, corr_bias, RPN);
  heads_softmax<<<gs(2 * NPIX, B), B, 0, stream>>>(RPN, w_cls, b_cls, w_box, b_box, out);
}
